// DynamicTimeGenerator_74423193305276
// MI455X (gfx1250) — matrix operand order repaired and checked by reading, not compiled, not run on hardware
//
#include <hip/hip_runtime.h>
#include <hip/hip_bf16.h>

// ---------------------------------------------------------------------------
// GRU-style dynamic time generator for MI455X (gfx1250, wave32, WMMA).
// B=256, H=768, T<=128.
// 64 workgroups = 16 batch tiles x 4 column quarters, 384 threads (12 waves).
// Each WG computes a (16 x 576) slice of the (16 x 2304) gate GEMM per step
// with v_wmma_f32_16x16x32_bf16, updates its 192 h-columns, and exchanges
// h slices with its 3 sibling WGs through L2 + an agent-scope flag barrier.
// Weights (bf16, packed to the WMMA B lane layout) stream from L2 each step.
// ---------------------------------------------------------------------------

typedef __attribute__((ext_vector_type(16))) __bf16          v16bf;
typedef __attribute__((ext_vector_type(8)))  float           v8f;
typedef __attribute__((ext_vector_type(16))) unsigned short  v16us;
typedef __attribute__((ext_vector_type(8)))  unsigned short  v8us;

#define HID      768
#define GATES3   2304          // 3*HID
#define BT       16            // batch rows per group
#define NG       4             // column-split workgroups per batch tile
#define NWAVES   12            // waves per WG
#define THREADS  384
#define KT       24            // 768 / 32 k-steps
#define TMAX     128
#define NBG      16            // batch groups (256/16)

__device__ __forceinline__ unsigned short f2bf(float f) {
    union { float f; unsigned int u; } x; x.f = f;
    unsigned int u = x.u;
    u += 0x7fffu + ((u >> 16) & 1u);   // round-to-nearest-even
    return (unsigned short)(u >> 16);
}
// Raw v_rcp_f32 (~1 ulp) instead of the precise-division expansion.
__device__ __forceinline__ float fast_rcp(float x) {
    return __builtin_amdgcn_rcpf(x);
}
__device__ __forceinline__ float fast_sigmoid(float x) {
    return fast_rcp(1.0f + __expf(-x));
}
__device__ __forceinline__ float fast_tanh(float x) {
    float e = __expf(2.0f * x);
    return (e - 1.0f) * fast_rcp(e + 1.0f);
}

union BV16 { v16us u; v16bf b; };

// Pack weight_hh (2304x768 f32) into bf16 tiles matching the WMMA B-matrix
// (32x16) lanes in this kernel's k order (same as A): lane l supplies N=l&15, K=(l>>4)*16 + j, contiguous.
// Tile index = tn*KT + tk ; per-tile 32 lanes x 16 ushorts (1 KB).
__global__ void pack_whh_kernel(const float* __restrict__ whh,
                                unsigned short* __restrict__ wpack) {
    int e = blockIdx.x * 256 + threadIdx.x;
    if (e >= GATES3 * HID) return;
    int tile = e >> 9;
    int lane = (e >> 4) & 31;
    int j    = e & 15;
    int tn = tile / KT, tk = tile % KT;
    int n = tn * 16 + (lane & 15);
    int k = tk * 32 + ((lane >> 4) << 4) + j;
    wpack[e] = f2bf(whh[n * HID + k]);
}

__global__ void reset_flags_kernel(int* __restrict__ flags, int n) {
    int i = blockIdx.x * 256 + threadIdx.x;
    if (i < n) flags[i] = 0;
}

__global__ __launch_bounds__(THREADS)
void gru_persistent_kernel(const float* __restrict__ context,
                           const float* __restrict__ bias_ih,
                           const float* __restrict__ bias_hh,
                           const float* __restrict__ fc_w,
                           const float* __restrict__ fc_b,
                           const int*   __restrict__ num_pred,
                           const unsigned short* __restrict__ wpack,
                           unsigned short* __restrict__ hglob,   // [NBG][BT][HID]
                           float* __restrict__ fcpart,           // [NBG][NG][T][BT]
                           int* __restrict__ flags,              // [NBG][T]
                           float* __restrict__ out,
                           int T) {
    __shared__ unsigned short sh_h[BT * HID];     // full h, bf16     (24 KB)
    __shared__ float sh_part[NWAVES * BT];        // fc partials      (768 B)
    __shared__ float sh_cum[TMAX * BT];           // cumsum (epilogue, 8 KB)

    const int tid = threadIdx.x;
    const int w   = tid >> 5;            // wave 0..11
    const int l   = tid & 31;
    const int lh  = l >> 4;
    const int lc  = l & 15;
    const int gb  = blockIdx.x >> 2;     // batch group 0..15
    const int gn  = blockIdx.x & 3;      // column quarter 0..3
    const int b0  = gb * BT;

    // This wave's 16 h-columns: tile tnW within [0,48).
    const int tnW = gn * NWAVES + w;
    const int col = tnW * 16 + lc;       // h column in [0,768)

    const float bsum_r = bias_ih[col]           + bias_hh[col];
    const float bsum_z = bias_ih[HID + col]     + bias_hh[HID + col];
    const float bihn   = bias_ih[2 * HID + col];
    const float bhhn   = bias_hh[2 * HID + col];
    const float fcw    = fc_w[col];
    const float fcb    = fc_b[0];

    // Per-gate weight tile element offsets (constant over t, tk).
    unsigned qoff[3];
#pragma unroll
    for (int g = 0; g < 3; ++g)
        qoff[g] = (unsigned)((g * 48 + tnW) * KT) * 512u + (unsigned)l * 16u;

    // ---- Init: full h(0) = context tile into LDS; own columns into regs ---
    for (int idx = tid; idx < BT * HID; idx += THREADS) {
        int m = idx / HID, k = idx % HID;
        sh_h[idx] = f2bf(context[(size_t)(b0 + m) * HID + k]);
    }
    v8f hreg;
#pragma unroll
    for (int v = 0; v < 8; ++v)
        hreg[v] = context[(size_t)(b0 + lh * 8 + v) * HID + col];
    __syncthreads();

    unsigned short* hg = hglob + (size_t)gb * BT * HID;

    for (int t = 0; t < T; ++t) {
        // Opaque zero offset: stop LICM from hoisting weight loads out of the
        // time loop (would spill); weights must re-stream from L2 each step.
        unsigned toff = 0;
        asm volatile("" : "+v"(toff));
        const unsigned short* wp = wpack + toff;

        // ---- GEMM: acc[g] += A(16x32) * B(32x16) over 24 k-steps ----------
        v8f acc[3];
#pragma unroll
        for (int g = 0; g < 3; ++g) acc[g] = 0.0f;

#pragma unroll 2
        for (int tk = 0; tk < KT; ++tk) {
            const unsigned short* ap = &sh_h[lc * HID + tk * 32 + lh * 16];
            v8us alo = *(const v8us*)(ap);
            v8us ahi = *(const v8us*)(ap + 8);
            BV16 abits;
            abits.u = __builtin_shufflevector(alo, ahi,
                        0, 1, 2, 3, 4, 5, 6, 7, 8, 9, 10, 11, 12, 13, 14, 15);
            v16bf A = abits.b;

            BV16 bt[3];
            const unsigned kadd = (unsigned)tk * 512u;
#pragma unroll
            for (int g = 0; g < 3; ++g)
                bt[g].u = *(const v16us*)(wp + (size_t)(qoff[g] + kadd));
#pragma unroll
            for (int g = 0; g < 3; ++g)
                acc[g] = __builtin_amdgcn_wmma_f32_16x16x32_bf16(
                    false, A, false, bt[g].b, (short)0, acc[g], false, false);
        }

        // ---- Gate fusion + fc partial -------------------------------------
        float partial[8];
#pragma unroll
        for (int v = 0; v < 8; ++v) {
            float r  = fast_sigmoid(acc[0][v] + bsum_r);
            float z  = fast_sigmoid(acc[1][v] + bsum_z);
            float nn = fast_tanh(bihn + r * (acc[2][v] + bhhn));
            float hn = (1.0f - z) * nn + z * hreg[v];
            hreg[v] = hn;
            partial[v] = hn * fcw;
        }
#pragma unroll
        for (int v = 0; v < 8; ++v) {
            partial[v] += __shfl_xor(partial[v], 1, 32);
            partial[v] += __shfl_xor(partial[v], 2, 32);
            partial[v] += __shfl_xor(partial[v], 4, 32);
            partial[v] += __shfl_xor(partial[v], 8, 32);
        }

        // ---- Publish h_new slice to L2; reduce fc partials in LDS ---------
#pragma unroll
        for (int v = 0; v < 8; ++v)
            hg[(lh * 8 + v) * HID + col] = f2bf(hreg[v]);
        if (lc == 0) {
#pragma unroll
            for (int v = 0; v < 8; ++v)
                sh_part[w * BT + lh * 8 + v] = partial[v];
        }
        __syncthreads();
        if (tid < BT) {
            float s = 0.0f;
#pragma unroll
            for (int ww = 0; ww < NWAVES; ++ww) s += sh_part[ww * BT + tid];
            fcpart[(((size_t)gb * NG + gn) * TMAX + t) * BT + tid] = s;
        }

        // ---- Group barrier over the 4 sibling WGs (through L2) ------------
        __threadfence();                       // release h + fcpart stores
        __syncthreads();
        if (tid == 0) {
            int* f = &flags[gb * TMAX + t];
            __hip_atomic_fetch_add(f, 1, __ATOMIC_ACQ_REL,
                                   __HIP_MEMORY_SCOPE_AGENT);
            while (__hip_atomic_load(f, __ATOMIC_ACQUIRE,
                                     __HIP_MEMORY_SCOPE_AGENT) < NG)
                __builtin_amdgcn_s_sleep(2);
        }
        __syncthreads();
        __threadfence();                       // acquire: invalidate near cache

        // ---- Refresh full h into LDS for the next step --------------------
#pragma unroll
        for (int i = 0; i < (BT * HID) / (8 * THREADS); ++i) {
            int chunk = tid + i * THREADS;     // 8-ushort (16 B) chunks
            *(v8us*)(&sh_h[chunk * 8]) = *(const v8us*)(&hg[chunk * 8]);
        }
        __syncthreads();
    }

    // ---- Epilogue (column quarter 0 only): softplus, cumsum, output -------
    if (gn != 0) return;

    if (tid < BT) {
        float cum = 0.0f;
        for (int t = 0; t < T; ++t) {
            float s = 0.0f;
#pragma unroll
            for (int q = 0; q < NG; ++q)
                s += fcpart[(((size_t)gb * NG + q) * TMAX + t) * BT + tid];
            float x = s + fcb;
            float d = (x > 20.0f) ? x : log1pf(__expf(x));  // softplus
            cum += d;
            sh_cum[t * BT + tid] = cum;
        }
    }
    __syncthreads();

    const int W = T + 2;
    for (int idx = tid; idx < BT * W; idx += THREADS) {
        int m = idx / W;
        int c = idx % W;
        int b = b0 + m;
        int nb = num_pred[b];
        int li = nb - 1; li = li < 0 ? 0 : (li > T - 1 ? T - 1 : li);
        float rlast = fast_rcp(sh_cum[li * BT + m] + 1e-6f);
        float val = 0.0f;
        if (c >= 1) {
            int tt = c - 1;
            if (tt < nb) val = sh_cum[tt * BT + m] * rlast;
        }
        if (c == nb + 1) val = 1.0f;
        out[(size_t)b * W + c] = val;
    }
}

extern "C" void kernel_launch(void* const* d_in, const int* in_sizes, int n_in,
                              void* d_out, int out_size, void* d_ws, size_t ws_size,
                              hipStream_t stream) {
    const float* context  = (const float*)d_in[0];
    // d_in[1] = weight_ih : unused by the reference recurrence
    const float* whh      = (const float*)d_in[2];
    const float* bias_ih  = (const float*)d_in[3];
    const float* bias_hh  = (const float*)d_in[4];
    const float* fc_w     = (const float*)d_in[5];
    const float* fc_b     = (const float*)d_in[6];
    const int*   num_pred = (const int*)d_in[7];

    const int B = in_sizes[0] / HID;            // 256
    int T = out_size / B - 2;                   // 128
    if (T > TMAX) T = TMAX;
    if (T < 1)    T = 1;

    // Workspace layout (≈4.5 MB total):
    unsigned short* wpack = (unsigned short*)d_ws;             // 3.54 MB
    unsigned short* hglob = wpack + (size_t)GATES3 * HID;      // 384 KB
    float* fcpart = (float*)(hglob + (size_t)NBG * BT * HID);  // 512 KB
    int*   flags  = (int*)(fcpart + (size_t)NBG * NG * TMAX * BT); // 8 KB

    const int total = GATES3 * HID;
    pack_whh_kernel<<<(total + 255) / 256, 256, 0, stream>>>(whh, wpack);

    const int nflags = NBG * TMAX;
    reset_flags_kernel<<<(nflags + 255) / 256, 256, 0, stream>>>(flags, nflags);

    gru_persistent_kernel<<<(B / BT) * NG, THREADS, 0, stream>>>(
        context, bias_ih, bias_hh, fc_w, fc_b, num_pred, wpack,
        hglob, fcpart, flags, (float*)d_out, T);
}
